// SearchTransfer_40261023433211
// MI455X (gfx1250) — matrix operand order repaired and checked by reading, not compiled, not run on hardware
//
#include <hip/hip_runtime.h>
#include <hip/hip_bf16.h>

typedef __attribute__((ext_vector_type(16))) _Float16 v16h;
typedef __attribute__((ext_vector_type(8)))  _Float16 v8h;
typedef __attribute__((ext_vector_type(8)))  float    v8f;

constexpr int B_   = 8;
constexpr int C_   = 64;
constexpr int H_   = 96;
constexpr int W_   = 96;
constexpr int HQ   = 47;              // (96+2-5)/2+1
constexpr int LQ   = HQ * HQ;         // 2209 query/key patches
constexpr int KF   = C_ * 25;         // 1600 feature dim
constexpr int LPAD = 2240;            // 140 tiles of 16 (>= LQ, multiple of 16*MT and 16*NT)
constexpr int MT   = 4;               // M-tiles (key rows) per wave
constexpr int NT   = 2;               // N-tiles (query cols) per wave  -> 64x32 block
constexpr int QG   = (LPAD / 16) / NT;// 70 column-tile groups
constexpr int KGRP = (LPAD / 16) / MT;// 35 row-tile groups
constexpr int C9   = C_ * 9;          // 576 value-patch channels

// ---- order-preserving float <-> u32 (for packed atomic max) ----
__device__ __forceinline__ unsigned int f2ord(float f) {
    unsigned int u = __float_as_uint(f);
    return (u & 0x80000000u) ? ~u : (u | 0x80000000u);
}
__device__ __forceinline__ float ord2f(unsigned int o) {
    unsigned int u = (o & 0x80000000u) ? (o & 0x7FFFFFFFu) : ~o;
    return __uint_as_float(u);
}

// ---- kernel 0: zero the packed (val,idx) argmax table ----
__global__ void zero_racc_kernel(unsigned long long* __restrict__ racc) {
    int t = blockIdx.x * blockDim.x + threadIdx.x;
    if (t < B_ * LPAD) racc[t] = 0ULL;
}

// ---- kernel 1: unfold(5,1,2) + L2-normalize + pack to f16 rows [B][LPAD][KF] ----
__global__ __launch_bounds__(32) void pack_norm_kernel(const float* __restrict__ src,
                                                       _Float16* __restrict__ dst) {
    const int row  = blockIdx.x;   // 0..LPAD-1
    const int b    = blockIdx.y;   // 0..B-1
    const int lane = threadIdx.x;  // 0..31
    _Float16* out = dst + ((size_t)b * LPAD + row) * KF;
    if (row >= LQ) {               // zero padding rows (never win the max; safe in WMMA)
        for (int i = lane; i < KF; i += 32) out[i] = (_Float16)0.0f;
        return;
    }
    const int ry = row / HQ, rx = row % HQ;
    const float* sb = src + (size_t)b * C_ * H_ * W_;
    float v[KF / 32];
    float ss = 0.0f;
#pragma unroll
    for (int i = 0; i < KF / 32; ++i) {
        int k  = lane + 32 * i;                 // 0..1599  (c*25 + kh*5 + kw)
        int c  = k / 25, r = k % 25;
        int kh = r / 5,  kw = r % 5;
        int iy = ry * 2 + kh - 1, ix = rx * 2 + kw - 1;
        bool ok = (iy >= 0) & (iy < H_) & (ix >= 0) & (ix < W_);
        int iyc = ok ? iy : 0, ixc = ok ? ix : 0;
        float x = sb[((size_t)c * H_ + iyc) * W_ + ixc];
        x = ok ? x : 0.0f;
        v[i] = x;
        ss += x * x;
    }
#pragma unroll
    for (int m = 16; m >= 1; m >>= 1) ss += __shfl_xor(ss, m, 32);
    float scale = 1.0f / fmaxf(sqrtf(ss), 1e-12f);
#pragma unroll
    for (int i = 0; i < KF / 32; ++i) out[lane + 32 * i] = (_Float16)(v[i] * scale);
}

// ---- kernel 2: WMMA correlation GEMM (64x32 block/wave) + streaming packed argmax ----
__global__ __launch_bounds__(32) void corr_argmax_kernel(
        const _Float16* __restrict__ kbuf,   // A source: key patches   [B][LPAD][KF]
        const _Float16* __restrict__ qbuf,   // B source: query patches [B][LPAD][KF]
        unsigned long long* __restrict__ racc) {
    const int lane = threadIdx.x;
    const int qg   = blockIdx.x;   // column tile group (queries), NT tiles
    const int kg   = blockIdx.y;   // row tile group (keys), MT tiles
    const int b    = blockIdx.z;
    const int n    = lane & 15;
    const int hi   = lane >> 4;

    const _Float16* qrow[NT];
#pragma unroll
    for (int ni = 0; ni < NT; ++ni)
        qrow[ni] = qbuf + ((size_t)b * LPAD + ((qg * NT + ni) * 16 + n)) * KF;
    const _Float16* kbase = kbuf + (size_t)b * LPAD * KF;

    v8f acc[MT][NT] = {};
    for (int ks = 0; ks < KF; ks += 32) {
        // B fragments (contiguous k, same as A): lane n holds K = ks + 16*hi + [0..15], N = n
        v16h bfrag[NT];
#pragma unroll
        for (int ni = 0; ni < NT; ++ni)
            bfrag[ni] = *(const v16h*)(qrow[ni] + ks + 16 * hi);
#pragma unroll
        for (int mi = 0; mi < MT; ++mi) {
            const int mrow = (kg * MT + mi) * 16 + n;    // A: M = lane%16
            const _Float16* arow = kbase + (size_t)mrow * KF + ks;
            // A fragment: K = +16*hi+[0..15] (matches B)
            v16h afrag = *(const v16h*)(arow + 16 * hi);
#pragma unroll
            for (int ni = 0; ni < NT; ++ni)
                acc[mi][ni] = __builtin_amdgcn_wmma_f32_16x16x32_f16(
                    false, afrag, false, bfrag[ni], (short)0, acc[mi][ni], false, false);
        }
    }

    // per column-group: lane-local best (m strictly increasing -> first-index ties),
    // then combine lane pairs (n, n+16) and one packed atomic per column.
#pragma unroll
    for (int ni = 0; ni < NT; ++ni) {
        float best = -3.0e38f;
        int   bidx = 0;
#pragma unroll
        for (int mi = 0; mi < MT; ++mi) {
#pragma unroll
            for (int v = 0; v < 8; ++v) {
                int m = (kg * MT + mi) * 16 + hi * 8 + v;    // D: M = v + 8*hi
                float val = acc[mi][ni][v];
                bool take = (m < LQ) && (val > best);
                best = take ? val : best;
                bidx = take ? m : bidx;
            }
        }
        float oval = __shfl_xor(best, 16, 32);
        int   oidx = __shfl_xor(bidx, 16, 32);
        if (oval > best || (oval == best && oidx < bidx)) { best = oval; bidx = oidx; }

        int q = (qg * NT + ni) * 16 + n;
        if (hi == 0 && q < LQ) {
            unsigned long long packed =
                ((unsigned long long)f2ord(best) << 32) |
                (unsigned long long)(0xFFFFFFFFu - (unsigned int)bidx);  // ~idx: first index wins ties
            atomicMax(racc + (size_t)b * LPAD + q, packed);
        }
    }
}

// ---- kernel 3: decode packed table -> S map + index array ----
__global__ void finalize_kernel(const unsigned long long* __restrict__ racc,
                                float* __restrict__ Sout, int* __restrict__ idxout) {
    int t = blockIdx.x * blockDim.x + threadIdx.x;
    if (t >= B_ * LQ) return;
    int b = t / LQ, q = t % LQ;
    unsigned long long p = racc[(size_t)b * LPAD + q];
    Sout[t]   = ord2f((unsigned int)(p >> 32));
    idxout[t] = (int)(0xFFFFFFFFu - (unsigned int)(p & 0xFFFFFFFFu));
}

// ---- kernel 4: gather unfold(value,3,1,1) columns by argmax index ----
__global__ void transfer_kernel(const float* __restrict__ value,
                                const int* __restrict__ idx,
                                float* __restrict__ T) {
    unsigned int t = blockIdx.x * blockDim.x + threadIdx.x;
    if (t >= (unsigned int)(B_ * C9 * LQ)) return;
    int q    = t % LQ;
    int rest = t / LQ;
    int c9   = rest % C9;
    int b    = rest / C9;
    int l  = idx[b * LQ + q];          // index along the 9216-long value-patch axis
    int y  = l / W_, x = l % W_;
    int c  = c9 / 9, r = c9 % 9;
    int kh = r / 3,  kw = r % 3;
    int iy = y + kh - 1, ix = x + kw - 1;
    float v = 0.0f;
    if (iy >= 0 && iy < H_ && ix >= 0 && ix < W_)
        v = value[(((size_t)b * C_ + c) * H_ + iy) * W_ + ix];
    T[t] = v;
}

extern "C" void kernel_launch(void* const* d_in, const int* in_sizes, int n_in,
                              void* d_out, int out_size, void* d_ws, size_t ws_size,
                              hipStream_t stream) {
    const float* queue = (const float*)d_in[0];
    const float* key   = (const float*)d_in[1];
    const float* value = (const float*)d_in[2];

    // workspace layout (bytes): kbuf | qbuf | racc | idx   (~115 MB total)
    char* ws = (char*)d_ws;
    const size_t buf_bytes = (size_t)B_ * LPAD * KF * sizeof(_Float16);   // 57,344,000
    _Float16* kbuf = (_Float16*)ws;
    _Float16* qbuf = (_Float16*)(ws + buf_bytes);
    unsigned long long* racc = (unsigned long long*)(ws + 2 * buf_bytes);
    int* idx = (int*)(ws + 2 * buf_bytes + (size_t)B_ * LPAD * sizeof(unsigned long long));

    float* Sout = (float*)d_out;              // [B,1,47,47]
    float* Tout = Sout + (size_t)B_ * LQ;     // [B,576,2209]

    dim3 gp(LPAD, B_);
    pack_norm_kernel<<<gp, 32, 0, stream>>>(key,   kbuf);
    pack_norm_kernel<<<gp, 32, 0, stream>>>(queue, qbuf);

    zero_racc_kernel<<<(B_ * LPAD + 255) / 256, 256, 0, stream>>>(racc);

    dim3 gg(QG, KGRP, B_);
    corr_argmax_kernel<<<gg, 32, 0, stream>>>(kbuf, qbuf, racc);

    finalize_kernel<<<(B_ * LQ + 255) / 256, 256, 0, stream>>>(racc, Sout, idx);

    unsigned int total = (unsigned int)(B_ * C9 * LQ);
    transfer_kernel<<<(total + 255) / 256, 256, 0, stream>>>(value, idx, Tout);
}
